// RecordMemory_52673478918498
// MI455X (gfx1250) — matrix operand order repaired and checked by reading, not compiled, not run on hardware
//
#include <hip/hip_runtime.h>
#include <hip/hip_bf16.h>

#define B_SZ 1024
#define D_SZ 2048
#define P_SZ 16384
#define C_SZ 8192
#define K_SZ 8
#define BG_KNN 50
#define NEG_V (-1e4f)
#define INV_TEMP 20.0f

typedef __attribute__((ext_vector_type(16))) _Float16 v16h;
typedef __attribute__((ext_vector_type(8)))  float    v8f;
typedef __attribute__((ext_vector_type(4)))  _Float16 v4h;
typedef __attribute__((ext_vector_type(4)))  int      v4i;

#if __has_builtin(__builtin_amdgcn_global_load_async_to_lds_b128) && \
    __has_builtin(__builtin_amdgcn_s_wait_asynccnt)
#define USE_ASYNC_LDS 1
#else
#define USE_ASYNC_LDS 0
#endif

// ---------------- helpers ----------------

__device__ inline unsigned f2key(float f) {
    unsigned u = __float_as_uint(f);
    return (u & 0x80000000u) ? ~u : (u | 0x80000000u);
}
__device__ inline float key2f(unsigned k) {
    return (k & 0x80000000u) ? __uint_as_float(k & 0x7FFFFFFFu)
                             : __uint_as_float(~k);
}

__device__ inline float block_reduce_sum(float v, volatile float* red) {
    const int tid = threadIdx.x;
    red[tid] = v; __syncthreads();
    for (int s = 128; s > 0; s >>= 1) {
        if (tid < s) red[tid] = red[tid] + red[tid + s];
        __syncthreads();
    }
    float r = red[0]; __syncthreads();
    return r;
}
__device__ inline float block_reduce_max(float v, volatile float* red) {
    const int tid = threadIdx.x;
    red[tid] = v; __syncthreads();
    for (int s = 128; s > 0; s >>= 1) {
        if (tid < s) red[tid] = fmaxf(red[tid], red[tid + s]);
        __syncthreads();
    }
    float r = red[0]; __syncthreads();
    return r;
}

#if USE_ASYNC_LDS
__device__ inline void async_cp16(const _Float16* g, _Float16* l) {
    __builtin_amdgcn_global_load_async_to_lds_b128(
        (__attribute__((address_space(1))) v4i*)(void*)g,
        (__attribute__((address_space(3))) v4i*)(void*)l,
        0, 0);
}
#endif

// ---------------- fp32 -> fp16 convert (vectorized) ----------------

__global__ __launch_bounds__(256)
void cvt_f32_f16_kernel(const float* __restrict__ in, _Float16* __restrict__ out, int n4) {
    int i = blockIdx.x * 256 + threadIdx.x;
    const int stride = gridDim.x * 256;
    for (; i < n4; i += stride) {
        const float4 v = reinterpret_cast<const float4*>(in)[i];
        v4h o = { (_Float16)v.x, (_Float16)v.y, (_Float16)v.z, (_Float16)v.w };
        reinterpret_cast<v4h*>(out)[i] = o;
    }
}

// ---------------- WMMA NT GEMM: C[M,N] = scale * A[M,K] * B[N,K]^T ----------------
// 128x128 block tile, 8 waves (wave32), each wave 32(M)x64(N) = 2x4 wmma tiles.
// K staged through LDS in chunks of 64 (two 16x16x32 wmma k-steps per stage).
// Async path: double-buffered LDS with GLOBAL_LOAD_ASYNC_TO_LDS_B128.

__device__ inline void wmma_stage(const _Float16* sA, const _Float16* sB,
                                  int lane, int wm, int wn, v8f (&acc)[2][4]) {
    #pragma unroll
    for (int kk = 0; kk < 64; kk += 32) {
        v16h a[2], b[4];
        // A fragment 16x32: lane<16 holds k 0-7 / 16-23, lane>=16 holds k 8-15 / 24-31
        #pragma unroll
        for (int mt = 0; mt < 2; ++mt) {
            const int m  = wm * 32 + mt * 16 + (lane & 15);
            const int kh = ((lane >> 4) & 1) * 8;
            const _Float16* p = &sA[m * 72 + kk + kh];
            *reinterpret_cast<uint4*>(&a[mt])       = *reinterpret_cast<const uint4*>(p);
            *(reinterpret_cast<uint4*>(&a[mt]) + 1) = *reinterpret_cast<const uint4*>(p + 16);
        }
        // B fragment 32x16: lane n = lane&15; lane<16 holds k 0-7 / 16-23, lane>=16 holds k 8-15 / 24-31
        #pragma unroll
        for (int nt = 0; nt < 4; ++nt) {
            const int n  = wn * 64 + nt * 16 + (lane & 15);
            const int kh = ((lane >> 4) & 1) * 8;
            const _Float16* p = &sB[n * 72 + kk + kh];
            *reinterpret_cast<uint4*>(&b[nt])       = *reinterpret_cast<const uint4*>(p);
            *(reinterpret_cast<uint4*>(&b[nt]) + 1) = *reinterpret_cast<const uint4*>(p + 16);
        }
        #pragma unroll
        for (int mt = 0; mt < 2; ++mt)
            #pragma unroll
            for (int nt = 0; nt < 4; ++nt)
                acc[mt][nt] = __builtin_amdgcn_wmma_f32_16x16x32_f16(
                    false, a[mt], false, b[nt], (short)0, acc[mt][nt], false, false);
    }
}

__global__ __launch_bounds__(256)
void gemm_nt_f16_kernel(const _Float16* __restrict__ A, const _Float16* __restrict__ Bm,
                        float* __restrict__ Cm, int N, int K, float scale) {
    const int tid  = threadIdx.x;
    const int lane = tid & 31;
    const int wave = tid >> 5;
    const int wm   = wave >> 1;                 // 0..3 -> M offset wm*32
    const int wn   = wave & 1;                  // 0..1 -> N offset wn*64
    const int m0   = blockIdx.y * 128;
    const int n0   = blockIdx.x * 128;

    v8f acc[2][4];
    #pragma unroll
    for (int i = 0; i < 2; ++i)
        #pragma unroll
        for (int j = 0; j < 4; ++j)
            #pragma unroll
            for (int r = 0; r < 8; ++r) acc[i][j][r] = 0.0f;

#if USE_ASYNC_LDS
    __shared__ __align__(16) _Float16 sA[2][128 * 72];
    __shared__ __align__(16) _Float16 sB[2][128 * 72];

    auto issue = [&](int buf, int kb) {
        #pragma unroll
        for (int it = 0; it < 4; ++it) {
            const int c   = tid + it * 256;
            const int row = c >> 3;
            const int kc  = (c & 7) << 3;
            async_cp16(&A[(size_t)(m0 + row) * K + kb + kc],  &sA[buf][row * 72 + kc]);
            async_cp16(&Bm[(size_t)(n0 + row) * K + kb + kc], &sB[buf][row * 72 + kc]);
        }
    };

    const int nst = K / 64;
    issue(0, 0);
    for (int i = 0; i < nst; ++i) {
        const int cur = i & 1;
        if (i + 1 < nst) {
            issue(cur ^ 1, (i + 1) * 64);                 // prefetch next stage
            __builtin_amdgcn_s_wait_asynccnt(8);          // stage i complete; 8 outstanding
        } else {
            __builtin_amdgcn_s_wait_asynccnt(0);
        }
        __syncthreads();
        wmma_stage(sA[cur], sB[cur], lane, wm, wn, acc);
        __syncthreads();                                  // buf reusable for stage i+2
    }
#else
    __shared__ __align__(16) _Float16 sA[128 * 72];
    __shared__ __align__(16) _Float16 sB[128 * 72];
    for (int kb = 0; kb < K; kb += 64) {
        #pragma unroll
        for (int it = 0; it < 4; ++it) {
            const int c   = tid + it * 256;
            const int row = c >> 3;
            const int kc  = (c & 7) << 3;
            *reinterpret_cast<uint4*>(&sA[row * 72 + kc]) =
                *reinterpret_cast<const uint4*>(&A[(size_t)(m0 + row) * K + kb + kc]);
            *reinterpret_cast<uint4*>(&sB[row * 72 + kc]) =
                *reinterpret_cast<const uint4*>(&Bm[(size_t)(n0 + row) * K + kb + kc]);
        }
        __syncthreads();
        wmma_stage(sA, sB, lane, wm, wn, acc);
        __syncthreads();
    }
#endif

    // Store: C/D layout -> VGPR r: M = r + 8*(lane>=16), N = lane&15
    const int rbase = 8 * (lane >> 4);
    const int nl    = lane & 15;
    #pragma unroll
    for (int mt = 0; mt < 2; ++mt) {
        #pragma unroll
        for (int nt = 0; nt < 4; ++nt) {
            const size_t base = (size_t)(m0 + wm * 32 + mt * 16 + rbase) * N
                              + (size_t)(n0 + wn * 64 + nt * 16 + nl);
            #pragma unroll
            for (int r = 0; r < 8; ++r)
                Cm[base + (size_t)r * N] = acc[mt][nt][r] * scale;
        }
    }
}

// ---------------- proxy-associate loss: per-row top-50 + log-softmax ----------------
// Row is staged (already masked) into LDS once; radix-select + exp passes run from LDS.

__global__ __launch_bounds__(256)
void proxy_loss_kernel(const float* __restrict__ scores,
                       const int* __restrict__ label2proxy,
                       const int* __restrict__ labels,
                       float* __restrict__ row_loss) {
    const int row = blockIdx.x;
    const int tid = threadIdx.x;
    const float* s = scores + (size_t)row * P_SZ;

    __shared__ __align__(16) float srow[P_SZ];   // 64 KB masked row cache
    __shared__ int      sp[K_SZ];
    __shared__ float    spos[K_SZ];
    __shared__ float    red[256];
    __shared__ unsigned hist[256];
    __shared__ unsigned s_prefix;
    __shared__ int      s_rem;

    if (tid < K_SZ) {
        const int p = label2proxy[labels[row] * K_SZ + tid];
        sp[tid]   = p;
        spos[tid] = (p >= 0) ? s[p] : NEG_V;   // gather positives BEFORE masking
    }
    if (tid == 0) { s_prefix = 0u; s_rem = BG_KNN; }
    __syncthreads();

    int   lsp[K_SZ];
    float lpos[K_SZ];
    #pragma unroll
    for (int k = 0; k < K_SZ; ++k) { lsp[k] = sp[k]; lpos[k] = spos[k]; }

    // Pass 1: stage masked row into LDS (masking applied once; duplicates accumulate
    // NEG like jax .at[].add) fused with the row-max reduction.
    float mx = -3.4e38f;
    for (int j4 = tid; j4 < P_SZ / 4; j4 += 256) {
        float4 v = reinterpret_cast<const float4*>(s)[j4];
        const int j = j4 * 4;
        #pragma unroll
        for (int k = 0; k < K_SZ; ++k) {
            if (lsp[k] == j + 0) v.x += NEG_V;
            if (lsp[k] == j + 1) v.y += NEG_V;
            if (lsp[k] == j + 2) v.z += NEG_V;
            if (lsp[k] == j + 3) v.w += NEG_V;
        }
        reinterpret_cast<float4*>(srow)[j4] = v;
        mx = fmaxf(mx, fmaxf(fmaxf(v.x, v.y), fmaxf(v.z, v.w)));
    }
    mx = block_reduce_max(mx, red);

    // Passes 2-5 (LDS): 8-bit radix select -> key of the 50th-largest masked score
    for (int shift = 24; shift >= 0; shift -= 8) {
        hist[tid] = 0u; __syncthreads();
        const unsigned pfx   = s_prefix;
        const unsigned pmask = (shift == 24) ? 0u : (0xFFFFFFFFu << (shift + 8));
        for (int j = tid; j < P_SZ; j += 256) {
            const unsigned key = f2key(srow[j]);
            if ((key & pmask) == pfx) atomicAdd(&hist[(key >> shift) & 255u], 1u);
        }
        __syncthreads();
        if (tid == 0) {
            int rem = s_rem;
            unsigned cum = 0;
            int b = 255;
            for (; b > 0; --b) {
                cum += hist[b];
                if ((int)cum >= rem) break;
            }
            if ((int)cum < rem) cum += hist[0];   // b == 0 fallthrough
            s_rem    = rem - (int)(cum - hist[b]);
            s_prefix = s_prefix | ((unsigned)b << shift);
        }
        __syncthreads();
    }
    const unsigned T  = s_prefix;
    const float    vT = key2f(T);

    // m = max over all 58 logits
    float mpos = NEG_V;
    #pragma unroll
    for (int k = 0; k < K_SZ; ++k) mpos = fmaxf(mpos, lpos[k]);
    const float m = fmaxf(mx, mpos);

    // Pass 6 (LDS): sum exp over strict top (key > T); ties filled analytically
    float sum = 0.0f, cgt = 0.0f;
    for (int j = tid; j < P_SZ; j += 256) {
        const float v = srow[j];
        if (f2key(v) > T) { sum += __expf(v - m); cgt += 1.0f; }
    }
    sum = block_reduce_sum(sum, red);
    cgt = block_reduce_sum(cgt, red);
    sum += ((float)BG_KNN - cgt) * __expf(vT - m);

    if (tid == 0) {
        float sp_sum = 0.0f;
        int   npos   = 0;
        #pragma unroll
        for (int k = 0; k < K_SZ; ++k) { sp_sum += __expf(lpos[k] - m); npos += (lsp[k] >= 0); }
        const float lse = m + __logf(sum + sp_sum);
        float accv = 0.0f;
        #pragma unroll
        for (int k = 0; k < K_SZ; ++k) if (lsp[k] >= 0) accv += (lse - lpos[k]);
        if (npos < 1) npos = 1;
        row_loss[row] = accv / (float)npos;
    }
}

// ---------------- class CE loss: single-pass online logsumexp over C ----------------

__global__ __launch_bounds__(256)
void class_loss_kernel(const float* __restrict__ scores,
                       const int* __restrict__ classes,
                       float* __restrict__ row_loss) {
    const int row = blockIdx.x;
    const int tid = threadIdx.x;
    const float* s = scores + (size_t)row * C_SZ;
    __shared__ float red_m[256];
    __shared__ float red_s[256];

    float m = -3.4e38f, sum = 0.0f;
    for (int j4 = tid; j4 < C_SZ / 4; j4 += 256) {
        const float4 v = reinterpret_cast<const float4*>(s)[j4];
        const float vv[4] = { v.x, v.y, v.z, v.w };
        #pragma unroll
        for (int c = 0; c < 4; ++c) {
            const float x = vv[c];
            if (x > m) { sum = sum * __expf(m - x) + 1.0f; m = x; }
            else       { sum += __expf(x - m); }
        }
    }
    red_m[tid] = m; red_s[tid] = sum; __syncthreads();
    for (int st = 128; st > 0; st >>= 1) {
        if (tid < st) {
            const float m1 = red_m[tid],      s1 = red_s[tid];
            const float m2 = red_m[tid + st], s2 = red_s[tid + st];
            const float mm = fmaxf(m1, m2);
            red_m[tid] = mm;
            red_s[tid] = s1 * __expf(m1 - mm) + s2 * __expf(m2 - mm);
        }
        __syncthreads();
    }
    if (tid == 0) {
        const float lse = red_m[0] + __logf(red_s[0]);
        row_loss[row] = lse - s[classes[row]];
    }
}

// ---------------- final scalar reduction ----------------

__global__ __launch_bounds__(256)
void final_reduce_kernel(const float* __restrict__ lp, const float* __restrict__ lc,
                         float* __restrict__ out) {
    __shared__ float red[256];
    const int tid = threadIdx.x;
    float a = 0.0f;
    for (int i = tid; i < B_SZ; i += 256) a += lp[i] + lc[i];
    a = block_reduce_sum(a, red);
    if (tid == 0) out[0] = a / (float)B_SZ;
}

// ---------------- launch ----------------

extern "C" void kernel_launch(void* const* d_in, const int* in_sizes, int n_in,
                              void* d_out, int out_size, void* d_ws, size_t ws_size,
                              hipStream_t stream) {
    const float* inputs        = (const float*)d_in[0];   // [B, D]
    const float* proxy_centers = (const float*)d_in[1];   // [P, D]
    const float* class_centers = (const float*)d_in[2];   // [C, D]
    const int*   label2proxy   = (const int*)d_in[3];     // [C, K]
    // d_in[4] = proxies : unused by the reference
    const int*   labels        = (const int*)d_in[5];     // [B]
    const int*   classes       = (const int*)d_in[6];     // [B]
    float* out = (float*)d_out;

    char*  ws  = (char*)d_ws;
    size_t off = 0;
    auto alloc = [&](size_t bytes) -> void* {
        void* p = ws + off;
        off += (bytes + 255) & ~(size_t)255;
        return p;
    };
    _Float16* inF16 = (_Float16*)alloc((size_t)B_SZ * D_SZ * 2);
    _Float16* pcF16 = (_Float16*)alloc((size_t)P_SZ * D_SZ * 2);
    _Float16* ccF16 = (_Float16*)alloc((size_t)C_SZ * D_SZ * 2);
    float*    scoresP = (float*)alloc((size_t)B_SZ * P_SZ * 4);
    float*    scoresC = (float*)alloc((size_t)B_SZ * C_SZ * 4);
    float*    lossP   = (float*)alloc((size_t)B_SZ * 4);
    float*    lossC   = (float*)alloc((size_t)B_SZ * 4);

    cvt_f32_f16_kernel<<<1024, 256, 0, stream>>>(inputs,        inF16, B_SZ * D_SZ / 4);
    cvt_f32_f16_kernel<<<4096, 256, 0, stream>>>(proxy_centers, pcF16, P_SZ * D_SZ / 4);
    cvt_f32_f16_kernel<<<4096, 256, 0, stream>>>(class_centers, ccF16, C_SZ * D_SZ / 4);

    gemm_nt_f16_kernel<<<dim3(P_SZ / 128, B_SZ / 128), 256, 0, stream>>>(
        inF16, pcF16, scoresP, P_SZ, D_SZ, INV_TEMP);
    gemm_nt_f16_kernel<<<dim3(C_SZ / 128, B_SZ / 128), 256, 0, stream>>>(
        inF16, ccF16, scoresC, C_SZ, D_SZ, INV_TEMP);

    proxy_loss_kernel<<<B_SZ, 256, 0, stream>>>(scoresP, label2proxy, labels, lossP);
    class_loss_kernel<<<B_SZ, 256, 0, stream>>>(scoresC, classes, lossC);
    final_reduce_kernel<<<1, 256, 0, stream>>>(lossP, lossC, out);

    (void)in_sizes; (void)n_in; (void)out_size; (void)ws_size;
}
